// GIN_5385888989902
// MI455X (gfx1250) — matrix operand order repaired and checked by reading, not compiled, not run on hardware
//
#include <hip/hip_runtime.h>
#include <hip/hip_bf16.h>
#include <stdint.h>

#define N_NODES 10000
#define N_EDGES 640000
#define F 128
#define M_TILES (N_NODES / 16)   // 625 16-row tiles

typedef __attribute__((ext_vector_type(16))) __bf16 v16bf;
typedef __attribute__((ext_vector_type(8)))  float  v8f;

__device__ __forceinline__ __bf16 to_bf16(float f) { return (__bf16)f; }

// ---------------------------------------------------------------------------
// h0 = x   (fuses the "+x" of GIN: scatter adds accumulate on top)
// ---------------------------------------------------------------------------
__global__ void k_init_h0(const float* __restrict__ x, float* __restrict__ h0) {
  int i = blockIdx.x * blockDim.x + threadIdx.x;
  if (i < N_NODES * F / 4) ((float4*)h0)[i] = ((const float4*)x)[i];
}

// ---------------------------------------------------------------------------
// Edge scatter: one wave per edge, lane handles 4 features.
// x (5.1MB) is L2-resident; atomics resolve in L2.
// ---------------------------------------------------------------------------
__global__ void k_scatter(const float* __restrict__ x,
                          const long long* __restrict__ ei,
                          float* __restrict__ h0) {
  const long long gid = (long long)blockIdx.x * blockDim.x + threadIdx.x;
  const long long edge = gid >> 5;
  if (edge >= N_EDGES) return;
  const int lane = threadIdx.x & 31;
  const long long src = ei[edge];
  const long long dst = ei[N_EDGES + edge];
  const float4 xv = *(const float4*)(x + src * F + lane * 4);
  float* dp = h0 + dst * F + lane * 4;
  atomicAdd(dp + 0, xv.x);
  atomicAdd(dp + 1, xv.y);
  atomicAdd(dp + 2, xv.z);
  atomicAdd(dp + 3, xv.w);
}

// ---------------------------------------------------------------------------
// Spectral norm: sigma = || W @ normalize(W^T @ u) ||  (exact identity)
// ---------------------------------------------------------------------------
__global__ void k_specnorm(const float* __restrict__ W, const float* __restrict__ u,
                           float* __restrict__ invSigma) {
  __shared__ float red[F];
  __shared__ float vsh[F];
  const int t = threadIdx.x;
  float v = 0.f;
  for (int k = 0; k < F; ++k) v += W[k * F + t] * u[k];   // v = W^T u
  red[t] = v * v;
  __syncthreads();
  for (int s = F / 2; s > 0; s >>= 1) { if (t < s) red[t] += red[t + s]; __syncthreads(); }
  const float vn = sqrtf(red[0]) + 1e-12f;
  vsh[t] = v / vn;
  __syncthreads();
  float wv = 0.f;
  for (int k = 0; k < F; ++k) wv += W[t * F + k] * vsh[k]; // W v
  red[t] = wv * wv;
  __syncthreads();
  for (int s = F / 2; s > 0; s >>= 1) { if (t < s) red[t] += red[t + s]; __syncthreads(); }
  if (t == 0) invSigma[0] = 1.0f / sqrtf(red[0]);
}

// ---------------------------------------------------------------------------
// Pre-swizzle W*(1/sigma) into bf16 WMMA B-fragments:
//   Wswz[((ks*8 + nt)*32 + lane)*16 + e] = bf16( W[o][kBase + half*8 + e + (e&8)] / sigma )
// with o = nt*16 + (lane&15), half = lane>>4, kBase = ks*32.
// 1024 threads, each writes one 32-byte fragment slice. Done ONCE; every GEMM
// wave then loads B with raw 128-bit loads (no per-wave conversion VALU).
// ---------------------------------------------------------------------------
__global__ void k_wswz(const float* __restrict__ W, const float* __restrict__ invSigma,
                       __bf16* __restrict__ Wswz) {
  const int t = blockIdx.x * blockDim.x + threadIdx.x;   // 0..1023
  if (t >= 4 * 8 * 32) return;
  const int lane = t & 31;
  const int nt   = (t >> 5) & 7;
  const int ks   = t >> 8;
  const int half = lane >> 4;
  const int l16  = lane & 15;
  const int o    = nt * 16 + l16;
  const float invS = invSigma[0];
  const float* wp = W + o * F + ks * 32 + half * 8;
  __bf16* dst = Wswz + t * 16;
#pragma unroll
  for (int e = 0; e < 16; ++e) dst[e] = to_bf16(wp[e + (e & 8)] * invS);
}

// ---------------------------------------------------------------------------
// BN stats + finalize
// ---------------------------------------------------------------------------
__global__ void k_bn_stats(const float* __restrict__ h1, float* __restrict__ stats) {
  const int f = threadIdx.x;              // blockDim.x == F
  float s = 0.f, sq = 0.f;
  for (int r = blockIdx.x; r < N_NODES; r += gridDim.x) {
    const float v = h1[r * F + f];
    s += v; sq += v * v;
  }
  atomicAdd(&stats[f], s);
  atomicAdd(&stats[F + f], sq);
}

__global__ void k_bn_finalize(const float* __restrict__ stats,
                              const float* __restrict__ gamma,
                              const float* __restrict__ beta,
                              float* __restrict__ scsh) {
  const int f = threadIdx.x;
  const float mu  = stats[f] * (1.0f / N_NODES);
  const float var = stats[F + f] * (1.0f / N_NODES) - mu * mu;
  const float inv = 1.0f / sqrtf(var + 1e-5f);
  const float sc  = gamma[f] * inv;
  scsh[f]     = sc;
  scsh[F + f] = beta[f] - mu * sc;
}

__global__ void k_zero(float* __restrict__ p, int n) {
  int i = blockIdx.x * blockDim.x + threadIdx.x;
  if (i < n) p[i] = 0.f;
}

// ---------------------------------------------------------------------------
// WMMA GEMM: Out = epi( A @ (W/sigma)^T + bias )
//   - one wave per 16-row M tile; 8 N-tiles x 4 K-steps = 32 v_wmma / wave
//   - B fragments come pre-converted/pre-swizzled from Wswz (raw v16bf loads)
//   - BN_IN: per-K affine on A before bf16 convert; RELU_OUT: relu after bias
// ---------------------------------------------------------------------------
template <bool BN_IN, bool RELU_OUT>
__global__ __launch_bounds__(256)
void gin_gemm_wmma(const float* __restrict__ A,        // N_NODES x F
                   const __bf16* __restrict__ Wswz,    // 4*8*32*16 bf16 fragments
                   const float* __restrict__ bias,     // F
                   const float* __restrict__ bnScale,  // [0,F): scale, [F,2F): shift
                   float* __restrict__ Out) {
  const int wavesPerBlock = blockDim.x >> 5;
  const int wid = blockIdx.x * wavesPerBlock + (threadIdx.x >> 5);
  if (wid >= M_TILES) return;                 // wave-uniform branch
  const int lane = threadIdx.x & 31;
  const int half = lane >> 4;
  const int l16  = lane & 15;
  const int row0 = wid * 16;

  v8f acc[8] = {};

#pragma unroll
  for (int ks = 0; ks < 4; ++ks) {            // K steps of 32
    const int kBase = ks * 32;
    // ---- A tile (16x32 bf16): lane l16 = row M; half picks K subgroup
    v16bf a;
    const float* ap = A + (row0 + l16) * F + kBase + half * 8;
#pragma unroll
    for (int e = 0; e < 8; ++e) {
      float f0 = ap[e];
      float f1 = ap[16 + e];
      if (BN_IN) {
        const int k0 = kBase + half * 8 + e;
        f0 = f0 * bnScale[k0]      + bnScale[F + k0];
        f1 = f1 * bnScale[k0 + 16] + bnScale[F + k0 + 16];
      }
      a[e]     = to_bf16(f0);     // K = kBase + half*8 + e
      a[8 + e] = to_bf16(f1);     // K = kBase + 16 + half*8 + e
    }
    // ---- 8 N-tiles: B fragment = two 128-bit loads, no conversion
#pragma unroll
    for (int nt = 0; nt < 8; ++nt) {
      const v16bf b = *(const v16bf*)(Wswz + (((ks << 3) + nt) * 32 + lane) * 16);
      acc[nt] = __builtin_amdgcn_wmma_f32_16x16x32_bf16(
          false, a, false, b, (short)0, acc[nt], false, false);
    }
  }

  // ---- epilogue: D 16x16 f32 layout (lane l16 = N, VGPR r = M = r + 8*half)
#pragma unroll
  for (int nt = 0; nt < 8; ++nt) {
    const int o = nt * 16 + l16;
    const float bv = bias[o];
#pragma unroll
    for (int r = 0; r < 8; ++r) {
      const int m = row0 + r + 8 * half;
      float v = acc[nt][r] + bv;
      if (RELU_OUT) v = fmaxf(v, 0.0f);
      Out[m * F + o] = v;
    }
  }
}

// ---------------------------------------------------------------------------
extern "C" void kernel_launch(void* const* d_in, const int* in_sizes, int n_in,
                              void* d_out, int out_size, void* d_ws, size_t ws_size,
                              hipStream_t stream) {
  const float*     x     = (const float*)d_in[0];
  const long long* ei    = (const long long*)d_in[1];   // int64 edge_index (2 x E)
  const float*     W1    = (const float*)d_in[2];
  const float*     b1    = (const float*)d_in[3];
  const float*     u1    = (const float*)d_in[4];
  const float*     gamma = (const float*)d_in[5];
  const float*     beta  = (const float*)d_in[6];
  const float*     W2    = (const float*)d_in[7];
  const float*     b2    = (const float*)d_in[8];
  const float*     u2    = (const float*)d_in[9];
  float*           out   = (float*)d_out;

  // workspace layout (float units; fragment buffers 32B-aligned)
  float* ws    = (float*)d_ws;
  float* h0    = ws;                       // N_NODES*F
  float* h1    = h0 + N_NODES * F;         // N_NODES*F
  float* stats = h1 + N_NODES * F;         // 2F
  float* scsh  = stats + 2 * F;            // 2F
  float* invS1 = scsh + 2 * F;             // 1
  float* invS2 = invS1 + 1;                // 1  (+6 pad to 32B boundary)
  __bf16* w1swz = (__bf16*)(invS2 + 7);    // 8192 dwords (16384 bf16)
  __bf16* w2swz = w1swz + 4 * 8 * 32 * 16;

  // 1) zero BN stats (re-zeroed every call: replay-safe)
  k_zero<<<1, 256, 0, stream>>>(stats, 2 * F);

  // 2) h0 = x
  {
    const int n4 = N_NODES * F / 4;
    k_init_h0<<<(n4 + 255) / 256, 256, 0, stream>>>(x, h0);
  }

  // 3) h0 += segment_sum(x[src] -> dst) : one wave per edge
  {
    const long long threads = (long long)N_EDGES * 32;
    const int blocks = (int)((threads + 255) / 256);
    k_scatter<<<blocks, 256, 0, stream>>>(x, ei, h0);
  }

  // 4) spectral norm scalars, then one-time bf16 weight swizzle
  k_specnorm<<<1, F, 0, stream>>>(W1, u1, invS1);
  k_specnorm<<<1, F, 0, stream>>>(W2, u2, invS2);
  k_wswz<<<4, 256, 0, stream>>>(W1, invS1, w1swz);
  k_wswz<<<4, 256, 0, stream>>>(W2, invS2, w2swz);

  // 5) h1 = relu(h0 @ (W1/sigma1)^T + b1)   [WMMA]
  {
    const int blocks = (M_TILES + 7) / 8;   // 8 waves / 256-thread block
    gin_gemm_wmma<false, true><<<blocks, 256, 0, stream>>>(h0, w1swz, b1, scsh, h1);
  }

  // 6) BN training stats over nodes
  k_bn_stats<<<256, F, 0, stream>>>(h1, stats);

  // 7) fold BN into per-feature affine
  k_bn_finalize<<<1, F, 0, stream>>>(stats, gamma, beta, scsh);

  // 8) out = BN(h1) @ (W2/sigma2)^T + b2   [WMMA, BN fused into A load]
  {
    const int blocks = (M_TILES + 7) / 8;
    gin_gemm_wmma<true, false><<<blocks, 256, 0, stream>>>(h1, w2swz, b2, scsh, out);
  }
}
